// NonlinearKernelTransformWithSkip_2551210574748
// MI455X (gfx1250) — matrix operand order repaired and checked by reading, not compiled, not run on hardware
//
#include <hip/hip_runtime.h>

typedef __attribute__((ext_vector_type(16))) _Float16 v16h;
typedef __attribute__((ext_vector_type(8)))  float    v8f;

#define WPB 4     // waves per block
#define NT  128   // threads per block

union Frag16 { v16h h; uint4 q[2]; };

// B fragment: N = ntile*16+(lane&15); elements 0..7  -> K = kc*32 + laneHi*8 + 0..7
//             elements 8..15 -> K = kc*32 + laneHi*8 + 16..23
// wt is stored transposed: wt[n*64 + k], k contiguous.
__device__ __forceinline__ v16h ldsB(const _Float16* wt, int ntile, int kc, int lane) {
  const int col = (ntile << 4) + (lane & 15);
  const int kb  = (kc << 5) + (((lane >> 4) & 1) << 3);
  const uint4* p = (const uint4*)(wt + (col << 6) + kb);
  Frag16 f; f.q[0] = p[0]; f.q[1] = p[2];
  return f.h;
}

// A fragment from LDS activation row (f16, 64 wide, k contiguous):
// elements 0..7  -> K = kc*32 + laneHi*8 + 0..7
// elements 8..15 -> K = kc*32 + laneHi*8 + 16..23
__device__ __forceinline__ v16h ldsA(const _Float16* hrow, int kc, int laneHi) {
  const int kb = (kc << 5) + (laneHi << 3);
  Frag16 f;
  f.q[0] = *(const uint4*)(hrow + kb);
  f.q[1] = *(const uint4*)(hrow + kb + 16);
  return f.h;
}

__global__ __launch_bounds__(NT)
void edge_mlp_kernel(const float* __restrict__ pos_i, const float* __restrict__ pos_j,
                     const float* __restrict__ v_j,  const float* __restrict__ x_j,
                     const float* __restrict__ W1, const float* __restrict__ b1,
                     const float* __restrict__ W2, const float* __restrict__ b2,
                     const float* __restrict__ W3, const float* __restrict__ b3,
                     float* __restrict__ out, int E, int ntiles)
{
  __shared__ __align__(16) _Float16 wt1[64 * 64];       //  8 KB (K padded 36->64)
  __shared__ __align__(16) _Float16 wt2[64 * 64];       //  8 KB
  __shared__ __align__(16) _Float16 wt3[272 * 64];      // 34 KB
  __shared__ __align__(16) _Float16 hbuf[WPB][16 * 64]; //  2 KB / wave
  __shared__ __align__(16) float    ctile[WPB][16 * 16];//  1 KB / wave

  const int tid    = threadIdx.x;
  const int lane   = tid & 31;
  const int wid    = tid >> 5;
  const int laneHi = (lane >> 4) & 1;
  const int l15    = lane & 15;

  // ---- stage weights (transposed, f16) into LDS ----
  for (int idx = tid; idx < 64 * 64; idx += NT) {
    const int n = idx >> 6, k = idx & 63;
    wt1[idx] = (k < 36) ? (_Float16)W1[k * 64 + n] : (_Float16)0.0f;
    wt2[idx] = (_Float16)W2[k * 64 + n];
  }
  for (int idx = tid; idx < 272 * 64; idx += NT) {
    const int n = idx >> 6, k = idx & 63;
    wt3[idx] = (_Float16)W3[k * 272 + n];
  }
  __syncthreads();

  // ---- hoist tile-invariant bias values into registers ----
  float b1v[4], b2v[4], b3v[17];
  #pragma unroll
  for (int n = 0; n < 4; ++n)  b1v[n] = b1[(n << 4) + l15];
  #pragma unroll
  for (int n = 0; n < 4; ++n)  b2v[n] = b2[(n << 4) + l15];
  #pragma unroll
  for (int n = 0; n < 17; ++n) b3v[n] = b3[(n << 4) + l15];

  _Float16* hrow = &hbuf[wid][l15 << 6];
  float*    ct   = ctile[wid];
  const int ctbase = (l15 << 4) + (laneHi << 3);   // [col][M] store base
  const int half = lane & 1;
  const int eL   = lane >> 1;                      // edge-in-tile for einsum

  const int waveGlobal = blockIdx.x * WPB + wid;
  const int nWaves     = gridDim.x * WPB;

  for (int tile = waveGlobal; tile < ntiles; tile += nWaves) {
    // ------------- build layer-1 A fragments directly from global -------------
    const int eA = min(tile * 16 + l15, E - 1);
    v16h A0, A1;
    #pragma unroll
    for (int i = 0; i < 16; ++i) A1[i] = (_Float16)0.0f;
    if (!laneHi) {
      const float2 pi = ((const float2*)pos_i)[eA];
      const float2 pj = ((const float2*)pos_j)[eA];
      const float4* vp = (const float4*)(v_j + (size_t)eA * 16);
      const float4* xp = (const float4*)(x_j + (size_t)eA * 16);
      const float4 va = vp[0], vd = vp[3], xa = xp[0], xd = xp[3];
      A0[0]=(_Float16)pi.x;  A0[1]=(_Float16)pi.y;  A0[2]=(_Float16)pj.x;  A0[3]=(_Float16)pj.y;
      A0[4]=(_Float16)va.x;  A0[5]=(_Float16)va.y;  A0[6]=(_Float16)va.z;  A0[7]=(_Float16)va.w;
      A0[8]=(_Float16)vd.x;  A0[9]=(_Float16)vd.y;  A0[10]=(_Float16)vd.z; A0[11]=(_Float16)vd.w;
      A0[12]=(_Float16)xa.x; A0[13]=(_Float16)xa.y; A0[14]=(_Float16)xa.z; A0[15]=(_Float16)xa.w;
      A1[0]=(_Float16)xd.x;  A1[1]=(_Float16)xd.y;  A1[2]=(_Float16)xd.z;  A1[3]=(_Float16)xd.w;
    } else {
      const float4* vp = (const float4*)(v_j + (size_t)eA * 16);
      const float4* xp = (const float4*)(x_j + (size_t)eA * 16);
      const float4 vb = vp[1], vc = vp[2], xb = xp[1], xc = xp[2];
      A0[0]=(_Float16)vb.x;  A0[1]=(_Float16)vb.y;  A0[2]=(_Float16)vb.z;  A0[3]=(_Float16)vb.w;
      A0[4]=(_Float16)vc.x;  A0[5]=(_Float16)vc.y;  A0[6]=(_Float16)vc.z;  A0[7]=(_Float16)vc.w;
      A0[8]=(_Float16)xb.x;  A0[9]=(_Float16)xb.y;  A0[10]=(_Float16)xb.z; A0[11]=(_Float16)xb.w;
      A0[12]=(_Float16)xc.x; A0[13]=(_Float16)xc.y; A0[14]=(_Float16)xc.z; A0[15]=(_Float16)xc.w;
    }

    // ------------- layer 1: 16x36(64) @ 64x64, relu -> hbuf -------------
    #pragma unroll
    for (int n = 0; n < 4; ++n) {
      v8f c = {};
      c = __builtin_amdgcn_wmma_f32_16x16x32_f16(false, A0, false, ldsB(wt1, n, 0, lane), (short)0, c, false, false);
      c = __builtin_amdgcn_wmma_f32_16x16x32_f16(false, A1, false, ldsB(wt1, n, 1, lane), (short)0, c, false, false);
      const int col = (n << 4) + l15;
      #pragma unroll
      for (int r = 0; r < 8; ++r) {
        float v = c[r] + b1v[n];
        v = v > 0.0f ? v : 0.0f;
        hbuf[wid][((r + (laneHi << 3)) << 6) + col] = (_Float16)v;
      }
    }
    asm volatile("s_wait_dscnt 0x0" ::: "memory");

    // ------------- layer 2: 16x64 @ 64x64, relu -> hbuf -------------
    const v16h A2_0 = ldsA(hrow, 0, laneHi);
    const v16h A2_1 = ldsA(hrow, 1, laneHi);
    asm volatile("" ::: "memory");
    #pragma unroll
    for (int n = 0; n < 4; ++n) {
      v8f c = {};
      c = __builtin_amdgcn_wmma_f32_16x16x32_f16(false, A2_0, false, ldsB(wt2, n, 0, lane), (short)0, c, false, false);
      c = __builtin_amdgcn_wmma_f32_16x16x32_f16(false, A2_1, false, ldsB(wt2, n, 1, lane), (short)0, c, false, false);
      const int col = (n << 4) + l15;
      #pragma unroll
      for (int r = 0; r < 8; ++r) {
        float v = c[r] + b2v[n];
        v = v > 0.0f ? v : 0.0f;
        hbuf[wid][((r + (laneHi << 3)) << 6) + col] = (_Float16)v;
      }
    }
    asm volatile("s_wait_dscnt 0x0" ::: "memory");

    // ------------- layer 3 + per-edge einsum -------------
    const v16h A3_0 = ldsA(hrow, 0, laneHi);
    const v16h A3_1 = ldsA(hrow, 1, laneHi);
    asm volatile("" ::: "memory");

    // v vector for the einsum: 2 lanes per edge
    const int eV = min(tile * 16 + eL, E - 1);
    const float4* vvp = (const float4*)(v_j + (size_t)eV * 16);
    const float4 t0 = vvp[0], t1 = vvp[1], t2 = vvp[2], t3 = vvp[3];
    float ve[16] = { t0.x,t0.y,t0.z,t0.w, t1.x,t1.y,t1.z,t1.w,
                     t2.x,t2.y,t2.z,t2.w, t3.x,t3.y,t3.z,t3.w };

    float oacc[8];
    #pragma unroll
    for (int i = 0; i < 8; ++i) oacc[i] = 0.0f;

    #pragma unroll
    for (int n = 0; n < 17; ++n) {
      v8f c = {};
      c = __builtin_amdgcn_wmma_f32_16x16x32_f16(false, A3_0, false, ldsB(wt3, n, 0, lane), (short)0, c, false, false);
      c = __builtin_amdgcn_wmma_f32_16x16x32_f16(false, A3_1, false, ldsB(wt3, n, 1, lane), (short)0, c, false, false);
      asm volatile("s_wait_dscnt 0x0" ::: "memory"); // prior ctile reads done before overwrite
      // packed [col][M] stage: 8 contiguous floats -> two b128 stores
      {
        const float bb = b3v[n];
        float4 s0 = { c[0] + bb, c[1] + bb, c[2] + bb, c[3] + bb };
        float4 s1 = { c[4] + bb, c[5] + bb, c[6] + bb, c[7] + bb };
        float4* cp = (float4*)(ct + ctbase);
        cp[0] = s0; cp[1] = s1;
      }
      asm volatile("s_wait_dscnt 0x0" ::: "memory");

      // columns 16n..16n+15 belong wholly to half 0 (n<8), wholly to half 1
      // (n>8), and split 8/8 at n==8 -> one divergent branch per n-tile.
      if (n < 8) {
        if (half == 0) {
          #pragma unroll
          for (int t = 0; t < 16; ++t) {
            const int colt = (n << 4) + t;
            const int o = colt / 17, cc = colt - o * 17;
            const float kv = ct[(t << 4) + eL];
            const float fv = (cc == 16) ? 1.0f : ve[cc];
            oacc[o & 7] += kv * fv;
          }
        }
      } else if (n > 8) {
        if (half == 1) {
          #pragma unroll
          for (int t = 0; t < 16; ++t) {
            const int colt = (n << 4) + t;
            const int o = colt / 17, cc = colt - o * 17;
            const float kv = ct[(t << 4) + eL];
            const float fv = (cc == 16) ? 1.0f : ve[cc];
            oacc[o & 7] += kv * fv;
          }
        }
      } else { // n == 8: t 0..7 -> o=7 (half 0), t 8..15 -> o=8 (half 1)
        if (half == 0) {
          #pragma unroll
          for (int t = 0; t < 8; ++t) {
            const int colt = 128 + t;
            const int cc = colt - 119;           // o = 7
            const float kv = ct[(t << 4) + eL];
            const float fv = (cc == 16) ? 1.0f : ve[cc];
            oacc[7] += kv * fv;
          }
        } else {
          #pragma unroll
          for (int t = 8; t < 16; ++t) {
            const int cc = t - 8;                // o = 8
            const float kv = ct[(t << 4) + eL];
            oacc[0] += kv * ve[cc];
          }
        }
      }
    }

    // ------------- store output -------------
    const int eOut = tile * 16 + eL;
    if (eOut < E) {
      float4 s0 = { oacc[0], oacc[1], oacc[2], oacc[3] };
      float4 s1 = { oacc[4], oacc[5], oacc[6], oacc[7] };
      float4* op = (float4*)(out + (size_t)eOut * 16 + (half << 3));
      op[0] = s0; op[1] = s1;
    }
  }
}

extern "C" void kernel_launch(void* const* d_in, const int* in_sizes, int n_in,
                              void* d_out, int out_size, void* d_ws, size_t ws_size,
                              hipStream_t stream) {
  const float* pos_i = (const float*)d_in[0];
  const float* pos_j = (const float*)d_in[1];
  const float* v_j   = (const float*)d_in[2];
  const float* x_j   = (const float*)d_in[3];
  const float* W1    = (const float*)d_in[4];
  const float* b1    = (const float*)d_in[5];
  const float* W2    = (const float*)d_in[6];
  const float* b2    = (const float*)d_in[7];
  const float* W3    = (const float*)d_in[8];
  const float* b3    = (const float*)d_in[9];
  float* out = (float*)d_out;

  const int E = in_sizes[0] / 2;   // pos_i is (E,2)
  if (E <= 0) return;
  const int ntiles = (E + 15) >> 4;
  int blocks = (ntiles + WPB - 1) / WPB;
  if (blocks > 2048) blocks = 2048;

  hipLaunchKernelGGL(edge_mlp_kernel, dim3(blocks), dim3(NT), 0, stream,
                     pos_i, pos_j, v_j, x_j, W1, b1, W2, b2, W3, b3, out, E, ntiles);
}
